// SkillVQVAE_86466281603563
// MI455X (gfx1250) — compile-verified, matrix operand order checked by reading, not run on hardware
//
#include <hip/hip_runtime.h>
#include <hip/hip_bf16.h>
#include <math.h>

typedef _Float16 half_t;
typedef __attribute__((ext_vector_type(16))) _Float16 v16h;
typedef __attribute__((ext_vector_type(8)))  _Float16 v8h;
typedef __attribute__((ext_vector_type(8)))  float    v8f;
typedef __attribute__((ext_vector_type(4)))  int      v4i;

#define BM 256
#define BN 64
#define BK 32

#if __has_builtin(__builtin_amdgcn_global_load_async_to_lds_b128) && \
    __has_builtin(__builtin_amdgcn_s_wait_asynccnt)
#define HAVE_ASYNC_LDS 1
typedef __attribute__((address_space(1))) v4i gas_v4i;   // global int4
typedef __attribute__((address_space(3))) v4i las_v4i;   // LDS int4
#else
#define HAVE_ASYNC_LDS 0
#endif

// ---------------------------------------------------------------------------
// WMMA GEMM: C[M,N] = act(alpha * (A @ Bw) + bias)
//   MODE 0: A is row-major [M,K] f16 (lda).
//   MODE 1: implicit im2col of conv1d k=4,s=2,p=1 over activation [B,Lin,C],
//           K index is k = kk*C + ci  (contiguous in ci -> 16B async chunks).
//   MODE 2: same for conv_transpose1d k=4,s=2,p=1 (lhs-dilated conv).
//   Bw f16 [K,N] row-major pre-packed to the same k ordering.
//   M must be a multiple of 256; K a multiple of 32.
//   Double-buffered LDS: tile k+1 streams in via ASYNCcnt DMA while tile k
//   feeds the WMMA pipe; one barrier per K-step.
// ---------------------------------------------------------------------------
struct GemmDesc {
  const half_t* A    = nullptr;
  const half_t* Bw   = nullptr;
  const float*  bias = nullptr;
  float*        Cf32 = nullptr;
  half_t*       Cf16 = nullptr;
  int M = 0, N = 0, K = 0;
  int lda = 0, ldc = 0;
  int loutShift = 0;   // log2(Lout)
  int Lin = 0;
  int cShift = 0;      // log2(C)
  float alpha = 1.0f;
  int act = 0;         // 0 none, 1 relu, 2 tanh
};

template <int MODE>
__global__ __launch_bounds__(256) void vqvae_wmma_gemm(GemmDesc g) {
  __shared__ __align__(16) half_t lA[2][BM][BK];  // [buf][m][k]
  __shared__ __align__(16) half_t lB[2][BN][BK];  // [buf][n][k] (transposed)

  const int tid  = threadIdx.x;
  const int lane = tid & 31;
  const int wave = tid >> 5;            // 8 waves, each owns a 32(M)x64(N) strip
  const int bm0  = blockIdx.x * BM;
  const int bn0  = blockIdx.y * BN;
  const int lrow = lane & 15;
  const int kbase = (lane >> 4) * 8;    // lanes 0-15: K{0..7,16..23}; 16-31: K{8..15,24..31}
  const int lmask = (1 << g.loutShift) - 1;
  const int cmask = (1 << g.cShift) - 1;
  const int nk = g.K / BK;

  // Stage tile kt into LDS buffer `buf` (A via async DMA, B via transpose).
  auto stage = [&](int kt, int buf) {
    const int k0 = kt * BK;
    if (kt + 1 < nk)
      __builtin_prefetch(g.Bw + (size_t)(k0 + BK) * g.N + bn0, 0, 3);

    // A tile: 256 rows x 32 k = 1024 16-byte chunks, 4 per thread.
#pragma unroll
    for (int it = 0; it < 4; ++it) {
      int slot = tid + it * 256;        // 0..1023
      int r  = slot >> 2;               // tile row 0..255
      int cc = (slot & 3) << 3;         // k offset 0,8,16,24
      int m = bm0 + r;
      int k = k0 + cc;
      const half_t* src = nullptr;
      if (MODE == 0) {
        src = &g.A[(size_t)m * g.lda + k];
      } else {
        int b  = m >> g.loutShift;
        int t  = m & lmask;
        int kk = k >> g.cShift;
        int ci = k & cmask;
        if (MODE == 1) {                // y[t] reads x[2t+kk-1]
          int ti = 2 * t + kk - 1;
          if (ti >= 0 && ti < g.Lin)
            src = &g.A[(((size_t)(b * g.Lin + ti)) << g.cShift) + ci];
        } else {                        // y[t] reads x[(t+1-kk)/2] when even
          int tp = t + 1 - kk;
          if (tp >= 0 && !(tp & 1)) {
            tp >>= 1;
            if (tp < g.Lin)
              src = &g.A[(((size_t)(b * g.Lin + tp)) << g.cShift) + ci];
          }
        }
      }
#if HAVE_ASYNC_LDS
      if (src) {
        __builtin_amdgcn_global_load_async_to_lds_b128(
            (gas_v4i*)src, (las_v4i*)&lA[buf][r][cc], 0, 0);
      } else {
        v8h z = {};
        *(v8h*)&lA[buf][r][cc] = z;
      }
#else
      v8h v = {};
      if (src) v = *(const v8h*)src;
      *(v8h*)&lA[buf][r][cc] = v;
#endif
    }

    // B tile: 32 k x 64 n, one v8h per thread, transposed to [n][k].
    {
      int kr = tid >> 3;                // 0..31
      int nc = (tid & 7) << 3;          // 0..56
      int n0 = bn0 + nc;
      v8h v = {};
      if (n0 < g.N) v = *(const v8h*)&g.Bw[(size_t)(k0 + kr) * g.N + n0];
#pragma unroll
      for (int q = 0; q < 8; ++q) lB[buf][nc + q][kr] = v[q];
    }
  };

  v8f acc[2][4] = {};

  // Prologue: stage tile 0, make it visible.
  stage(0, 0);
#if HAVE_ASYNC_LDS
  __builtin_amdgcn_s_wait_asynccnt(0);
#endif
  __syncthreads();

  for (int kt = 0; kt < nk; ++kt) {
    const int cur = kt & 1;

    // Kick off DMA of the next tile into the other buffer; it overlaps the
    // WMMA work below and is fenced by the single wait+barrier at loop end.
    if (kt + 1 < nk) stage(kt + 1, cur ^ 1);

    // ---- fragments + 8 WMMAs per wave ----
    v16h af[2], bf[4];
#pragma unroll
    for (int i = 0; i < 2; ++i) {
      int r = wave * 32 + i * 16 + lrow;
      v8h lo = *(const v8h*)&lA[cur][r][kbase];
      v8h hi = *(const v8h*)&lA[cur][r][16 + kbase];
      af[i] = __builtin_shufflevector(lo, hi, 0,1,2,3,4,5,6,7,8,9,10,11,12,13,14,15);
    }
#pragma unroll
    for (int j = 0; j < 4; ++j) {
      int c = j * 16 + lrow;
      v8h lo = *(const v8h*)&lB[cur][c][kbase];
      v8h hi = *(const v8h*)&lB[cur][c][16 + kbase];
      bf[j] = __builtin_shufflevector(lo, hi, 0,1,2,3,4,5,6,7,8,9,10,11,12,13,14,15);
    }
#pragma unroll
    for (int i = 0; i < 2; ++i)
#pragma unroll
      for (int j = 0; j < 4; ++j)
        acc[i][j] = __builtin_amdgcn_wmma_f32_16x16x32_f16(
            false, af[i], false, bf[j], (short)0, acc[i][j], false, false);

#if HAVE_ASYNC_LDS
    __builtin_amdgcn_s_wait_asynccnt(0);
#endif
    __syncthreads();
  }

  // ---- epilogue ----
  const int rowHi = (lane >> 4) * 8;
#pragma unroll
  for (int i = 0; i < 2; ++i) {
#pragma unroll
    for (int j = 0; j < 4; ++j) {
      int col = bn0 + j * 16 + lrow;
      if (col < g.N) {
        float b = g.bias ? g.bias[col] : 0.0f;
#pragma unroll
        for (int r = 0; r < 8; ++r) {
          int row = bm0 + wave * 32 + i * 16 + r + rowHi;
          float v = g.alpha * acc[i][j][r] + b;
          if (g.act == 1) v = fmaxf(v, 0.0f);
          else if (g.act == 2) v = tanhf(v);
          size_t o = (size_t)row * g.ldc + col;
          if (g.Cf32) g.Cf32[o] = v;
          if (g.Cf16) g.Cf16[o] = (half_t)v;
        }
      }
    }
  }
}

// ---------------------------------------------------------------------------
// Packing / reduction kernels
// ---------------------------------------------------------------------------
__global__ void vq_pack_f16(const float* src, half_t* dst, size_t n) {
  size_t stride = (size_t)gridDim.x * blockDim.x;
  for (size_t i = (size_t)blockIdx.x * blockDim.x + threadIdx.x; i < n; i += stride)
    dst[i] = (half_t)src[i];
}

// generic [N][K] f32 -> [K][N] f16 (used for codebook transpose)
__global__ void vq_transpose_pack(const float* src, half_t* dst, int N, int K) {
  size_t n = (size_t)N * K;
  size_t stride = (size_t)gridDim.x * blockDim.x;
  for (size_t i = (size_t)blockIdx.x * blockDim.x + threadIdx.x; i < n; i += stride) {
    int nn = (int)(i / K), k = (int)(i % K);
    dst[(size_t)k * N + nn] = (half_t)src[i];
  }
}

// conv weight [co][ci][kk] -> [(kk*Cin+ci)][co]
__global__ void vq_conv_w_pack(const float* src, half_t* dst, int Cin, int Cout, size_t n) {
  size_t stride = (size_t)gridDim.x * blockDim.x;
  for (size_t i = (size_t)blockIdx.x * blockDim.x + threadIdx.x; i < n; i += stride) {
    int kk = (int)(i & 3);
    size_t t = i >> 2;                 // co*Cin + ci
    int ci = (int)(t % Cin);
    int co = (int)(t / Cin);
    dst[((size_t)kk * Cin + ci) * Cout + co] = (half_t)src[i];
  }
}

// tconv weight [ci][co][kk] -> [(kk*Cin+ci)][co]
__global__ void vq_tconv_pack(const float* src, half_t* dst, int Cin, int Cout, size_t n) {
  size_t stride = (size_t)gridDim.x * blockDim.x;
  for (size_t i = (size_t)blockIdx.x * blockDim.x + threadIdx.x; i < n; i += stride) {
    int kk = (int)(i & 3);
    size_t t = i >> 2;                 // ci*Cout + co
    int co = (int)(t % Cout);
    int ci = (int)(t / Cout);
    dst[((size_t)kk * Cin + ci) * Cout + co] = (half_t)src[i];
  }
}

// exp_w [k][h*8+t] -> [k][t*1024+h] so g lands in [B,T8,H] layout
__global__ void vq_exp_pack(const float* src, half_t* dst) {
  size_t n = (size_t)256 * 8192;
  size_t stride = (size_t)gridDim.x * blockDim.x;
  for (size_t i = (size_t)blockIdx.x * blockDim.x + threadIdx.x; i < n; i += stride) {
    size_t k = i >> 13;
    int col = (int)(i & 8191);
    int h = col >> 3, t = col & 7;
    dst[(k << 13) + ((size_t)t << 10) + h] = (half_t)src[i];
  }
}

// proj_w [(d*8+t)][n] f32 -> dst [(t*256+d)][n] f16
__global__ void vq_proj_pack_kernel(const float* src, half_t* dst) {
  size_t n = (size_t)2048 * 256;
  size_t stride = (size_t)gridDim.x * blockDim.x;
  for (size_t i = (size_t)blockIdx.x * blockDim.x + threadIdx.x; i < n; i += stride) {
    int row = (int)(i >> 8), c = (int)(i & 255);
    int t = row >> 8, d = row & 255;
    dst[i] = (half_t)src[(size_t)(d * 8 + t) * 256 + c];
  }
}

__global__ void vq_cbnorm(const float* cb, float* cbn) {
  int r = blockIdx.x * blockDim.x + threadIdx.x;
  if (r < 4096) {
    float s = 0.0f;
    for (int d = 0; d < 256; ++d) { float v = cb[(size_t)r * 256 + d]; s += v * v; }
    cbn[r] = s;
  }
}

__global__ void vq_argmin(const float* S, int* idx, float* idxF) {
  __shared__ float sv[256];
  __shared__ int   si[256];
  int row = blockIdx.x;
  const float* s = S + (size_t)row * 4096;
  float best = 3.0e38f; int bi = 0;
  for (int j = threadIdx.x; j < 4096; j += 256) {
    float v = s[j];
    if (v < best) { best = v; bi = j; }
  }
  sv[threadIdx.x] = best; si[threadIdx.x] = bi;
  __syncthreads();
  for (int o = 128; o > 0; o >>= 1) {
    if (threadIdx.x < o) {
      float ov = sv[threadIdx.x + o]; int oi = si[threadIdx.x + o];
      if (ov < sv[threadIdx.x] || (ov == sv[threadIdx.x] && oi < si[threadIdx.x])) {
        sv[threadIdx.x] = ov; si[threadIdx.x] = oi;
      }
    }
    __syncthreads();
  }
  if (threadIdx.x == 0) { idx[row] = si[0]; idxF[row] = (float)si[0]; }
}

__global__ void vq_gather_zq(const int* idx, const float* cb, const half_t* cbh,
                             float* zqf, half_t* zqh) {
  size_t n = (size_t)2048 * 256;
  size_t stride = (size_t)gridDim.x * blockDim.x;
  for (size_t i = (size_t)blockIdx.x * blockDim.x + threadIdx.x; i < n; i += stride) {
    int b = (int)(i >> 8), d = (int)(i & 255);
    int k = idx[b];
    zqf[i] = cb[(size_t)k * 256 + d];
    zqh[i] = cbh[(size_t)k * 256 + d];
  }
}

__global__ void vq_znorm_pack(const float* z, half_t* zn, half_t* znT) {
  __shared__ float sh[256];
  int b = blockIdx.x, d = threadIdx.x;
  float v = z[(size_t)b * 256 + d];
  sh[d] = v * v;
  __syncthreads();
  for (int o = 128; o > 0; o >>= 1) { if (d < o) sh[d] += sh[d + o]; __syncthreads(); }
  float inv = rsqrtf(sh[0] + 1e-30f);
  half_t h = (half_t)(v * inv);
  zn[(size_t)b * 256 + d] = h;
  znT[(size_t)d * 2048 + b] = h;
}

__global__ void vq_init_accum(float* a) {
  if (threadIdx.x < 8) a[threadIdx.x] = 0.0f;
}

__global__ void vq_sqdiff_reduce(const float* a, const float* b, size_t n, float* accum) {
  __shared__ float sh[256];
  float s = 0.0f;
  size_t stride = (size_t)gridDim.x * blockDim.x;
  for (size_t i = (size_t)blockIdx.x * blockDim.x + threadIdx.x; i < n; i += stride) {
    float d = a[i] - b[i]; s += d * d;
  }
  sh[threadIdx.x] = s;
  __syncthreads();
  for (int o = 128; o > 0; o >>= 1) { if (threadIdx.x < o) sh[threadIdx.x] += sh[threadIdx.x + o]; __syncthreads(); }
  if (threadIdx.x == 0) atomicAdd(accum, sh[0]);
}

__global__ void vq_rot_reduce(const float* G, float* accum) {
  __shared__ float sh[256];
  size_t n = (size_t)2048 * 2048;
  float s = 0.0f;
  size_t stride = (size_t)gridDim.x * blockDim.x;
  for (size_t i = (size_t)blockIdx.x * blockDim.x + threadIdx.x; i < n; i += stride) {
    int r = (int)(i >> 11), c = (int)(i & 2047);
    if (r != c) { float v = G[i]; s += v * v; }
  }
  sh[threadIdx.x] = s;
  __syncthreads();
  for (int o = 128; o > 0; o >>= 1) { if (threadIdx.x < o) sh[threadIdx.x] += sh[threadIdx.x + o]; __syncthreads(); }
  if (threadIdx.x == 0) atomicAdd(accum, sh[0]);
}

__global__ void vq_finalize_loss(const float* accum, float* out) {
  float recon = accum[0] / (2048.0f * 64.0f * 32.0f);
  float vq    = 1.25f * accum[1] / (2048.0f * 256.0f);   // commit(0.25)+codebook(1.0)
  float rot   = 0.1f  * accum[2] / (2048.0f * 2048.0f);
  out[0] = recon + vq + rot;
}

// ---------------------------------------------------------------------------
// Host orchestration
// ---------------------------------------------------------------------------
extern "C" void kernel_launch(void* const* d_in, const int* in_sizes, int n_in,
                              void* d_out, int out_size, void* d_ws, size_t ws_size,
                              hipStream_t stream) {
  (void)in_sizes; (void)n_in; (void)ws_size;
  const float* action   = (const float*)d_in[0];
  const float* enc_w1   = (const float*)d_in[1];
  const float* enc_b1   = (const float*)d_in[2];
  const float* enc_w2   = (const float*)d_in[3];
  const float* enc_b2   = (const float*)d_in[4];
  const float* enc_w3   = (const float*)d_in[5];
  const float* enc_b3   = (const float*)d_in[6];
  const float* proj_w   = (const float*)d_in[7];
  const float* proj_b   = (const float*)d_in[8];
  const float* codebook = (const float*)d_in[9];
  const float* exp_w    = (const float*)d_in[10];
  const float* exp_b    = (const float*)d_in[11];
  const float* dec_w1   = (const float*)d_in[12];
  const float* dec_b1   = (const float*)d_in[13];
  const float* dec_w2   = (const float*)d_in[14];
  const float* dec_b2   = (const float*)d_in[15];
  const float* dec_w3   = (const float*)d_in[16];
  const float* dec_b3   = (const float*)d_in[17];
  float* out = (float*)d_out;

  char* ws = (char*)d_ws;
  size_t off = 0;
  auto alloc = [&](size_t bytes) -> void* {
    void* p = ws + off;
    off += (bytes + 255) & ~(size_t)255;
    return p;
  };
  const size_t H2 = sizeof(half_t);

  half_t* xa    = (half_t*)alloc((size_t)4194304 * H2);        // [B,64,32] f16
  half_t* w1p   = (half_t*)alloc((size_t)128 * 1024 * H2);
  half_t* w2p   = (half_t*)alloc((size_t)4096 * 1024 * H2);
  half_t* w3p   = (half_t*)alloc((size_t)4096 * 256 * H2);
  half_t* projp = (half_t*)alloc((size_t)2048 * 256 * H2);
  half_t* cbT   = (half_t*)alloc((size_t)256 * 4096 * H2);
  half_t* cbh   = (half_t*)alloc((size_t)4096 * 256 * H2);
  float*  cbn   = (float*) alloc((size_t)4096 * 4);
  half_t* expp  = (half_t*)alloc((size_t)256 * 8192 * H2);
  half_t* d1p   = (half_t*)alloc((size_t)4096 * 1024 * H2);
  half_t* d2p   = (half_t*)alloc((size_t)4096 * 1024 * H2);
  half_t* d3p   = (half_t*)alloc((size_t)4096 * 32 * H2);
  half_t* h1    = (half_t*)alloc((size_t)65536 * 1024 * H2);   // [B,32,1024]
  half_t* h2    = (half_t*)alloc((size_t)32768 * 1024 * H2);   // [B,16,1024]
  half_t* h3    = (half_t*)alloc((size_t)16384 * 256 * H2);    // [B,8,256]
  float*  zf    = (float*) alloc((size_t)2048 * 256 * 4);
  half_t* zh    = (half_t*)alloc((size_t)2048 * 256 * H2);
  float*  Smat  = (float*) alloc((size_t)2048 * 4096 * 4);
  int*    idxb  = (int*)   alloc((size_t)2048 * 4);
  half_t* zqh   = (half_t*)alloc((size_t)2048 * 256 * H2);
  float*  zqf   = (float*) alloc((size_t)2048 * 256 * 4);
  half_t* gbuf  = (half_t*)alloc((size_t)2048 * 8192 * H2);    // [B,8,1024]
  half_t* g1    = (half_t*)alloc((size_t)32768 * 1024 * H2);   // [B,16,1024]
  half_t* g2    = (half_t*)alloc((size_t)65536 * 1024 * H2);   // [B,32,1024]
  half_t* znh   = (half_t*)alloc((size_t)2048 * 256 * H2);
  half_t* znT   = (half_t*)alloc((size_t)256 * 2048 * H2);
  float*  Gmat  = (float*) alloc((size_t)2048 * 2048 * 4);
  float*  accum = (float*) alloc(8 * 4);

  auto grid_of = [](const GemmDesc& gd) {
    return dim3((unsigned)(gd.M / BM), (unsigned)((gd.N + BN - 1) / BN));
  };
  auto gemm0 = [&](const GemmDesc& gd) { vqvae_wmma_gemm<0><<<grid_of(gd), 256, 0, stream>>>(gd); };
  auto gemm1 = [&](const GemmDesc& gd) { vqvae_wmma_gemm<1><<<grid_of(gd), 256, 0, stream>>>(gd); };
  auto gemm2 = [&](const GemmDesc& gd) { vqvae_wmma_gemm<2><<<grid_of(gd), 256, 0, stream>>>(gd); };
  auto ew = [&](size_t n) {
    size_t b = (n + 255) / 256;
    return dim3((unsigned)(b < 16384 ? b : 16384));
  };

  // ---- packing ----
  vq_pack_f16<<<ew(4194304), 256, 0, stream>>>(action, xa, 4194304);
  vq_conv_w_pack<<<ew(131072), 256, 0, stream>>>(enc_w1, w1p, 32, 1024, 131072);
  vq_conv_w_pack<<<ew(4194304), 256, 0, stream>>>(enc_w2, w2p, 1024, 1024, 4194304);
  vq_conv_w_pack<<<ew(1048576), 256, 0, stream>>>(enc_w3, w3p, 1024, 256, 1048576);
  vq_proj_pack_kernel<<<ew(524288), 256, 0, stream>>>(proj_w, projp);
  vq_exp_pack<<<ew(2097152), 256, 0, stream>>>(exp_w, expp);
  vq_transpose_pack<<<ew(1048576), 256, 0, stream>>>(codebook, cbT, 4096, 256);
  vq_pack_f16<<<ew(1048576), 256, 0, stream>>>(codebook, cbh, 1048576);
  vq_cbnorm<<<16, 256, 0, stream>>>(codebook, cbn);
  vq_tconv_pack<<<ew(4194304), 256, 0, stream>>>(dec_w1, d1p, 1024, 1024, 4194304);
  vq_tconv_pack<<<ew(4194304), 256, 0, stream>>>(dec_w2, d2p, 1024, 1024, 4194304);
  vq_tconv_pack<<<ew(131072), 256, 0, stream>>>(dec_w3, d3p, 1024, 32, 131072);
  vq_init_accum<<<1, 32, 0, stream>>>(accum);

  // ---- encoder ----
  GemmDesc gd{};
  gd = GemmDesc{}; gd.A = xa; gd.Bw = w1p; gd.bias = enc_b1; gd.Cf16 = h1;
  gd.M = 65536; gd.N = 1024; gd.K = 128; gd.ldc = 1024;
  gd.loutShift = 5; gd.Lin = 64; gd.cShift = 5; gd.act = 1; gemm1(gd);

  gd = GemmDesc{}; gd.A = h1; gd.Bw = w2p; gd.bias = enc_b2; gd.Cf16 = h2;
  gd.M = 32768; gd.N = 1024; gd.K = 4096; gd.ldc = 1024;
  gd.loutShift = 4; gd.Lin = 32; gd.cShift = 10; gd.act = 1; gemm1(gd);

  gd = GemmDesc{}; gd.A = h2; gd.Bw = w3p; gd.bias = enc_b3; gd.Cf16 = h3;
  gd.M = 16384; gd.N = 256; gd.K = 4096; gd.ldc = 256;
  gd.loutShift = 3; gd.Lin = 16; gd.cShift = 10; gemm1(gd);

  gd = GemmDesc{}; gd.A = h3; gd.Bw = projp; gd.bias = proj_b; gd.Cf32 = zf; gd.Cf16 = zh;
  gd.M = 2048; gd.N = 256; gd.K = 2048; gd.lda = 2048; gd.ldc = 256; gemm0(gd);

  // ---- VQ: scores = ||c||^2 - 2 z.c ----
  gd = GemmDesc{}; gd.A = zh; gd.Bw = cbT; gd.bias = cbn; gd.Cf32 = Smat;
  gd.M = 2048; gd.N = 4096; gd.K = 256; gd.lda = 256; gd.ldc = 4096;
  gd.alpha = -2.0f; gemm0(gd);

  vq_argmin<<<2048, 256, 0, stream>>>(Smat, idxb, out + 4194304);
  vq_gather_zq<<<ew(524288), 256, 0, stream>>>(idxb, codebook, cbh, zqf, zqh);
  vq_sqdiff_reduce<<<2048, 256, 0, stream>>>(zqf, zf, (size_t)2048 * 256, accum + 1);

  // ---- rotation loss ----
  vq_znorm_pack<<<2048, 256, 0, stream>>>(zf, znh, znT);
  gd = GemmDesc{}; gd.A = znh; gd.Bw = znT; gd.Cf32 = Gmat;
  gd.M = 2048; gd.N = 2048; gd.K = 256; gd.lda = 256; gd.ldc = 2048; gemm0(gd);
  vq_rot_reduce<<<2048, 256, 0, stream>>>(Gmat, accum + 2);

  // ---- decoder (straight-through: input is z_q) ----
  gd = GemmDesc{}; gd.A = zqh; gd.Bw = expp; gd.bias = exp_b; gd.Cf16 = gbuf;
  gd.M = 2048; gd.N = 8192; gd.K = 256; gd.lda = 256; gd.ldc = 8192; gemm0(gd);

  gd = GemmDesc{}; gd.A = gbuf; gd.Bw = d1p; gd.bias = dec_b1; gd.Cf16 = g1;
  gd.M = 32768; gd.N = 1024; gd.K = 4096; gd.ldc = 1024;
  gd.loutShift = 4; gd.Lin = 8; gd.cShift = 10; gd.act = 1; gemm2(gd);

  gd = GemmDesc{}; gd.A = g1; gd.Bw = d2p; gd.bias = dec_b2; gd.Cf16 = g2;
  gd.M = 65536; gd.N = 1024; gd.K = 4096; gd.ldc = 1024;
  gd.loutShift = 5; gd.Lin = 16; gd.cShift = 10; gd.act = 1; gemm2(gd);

  gd = GemmDesc{}; gd.A = g2; gd.Bw = d3p; gd.bias = dec_b3; gd.Cf32 = out;
  gd.M = 131072; gd.N = 32; gd.K = 4096; gd.ldc = 32;
  gd.loutShift = 6; gd.Lin = 32; gd.cShift = 10; gd.act = 2; gemm2(gd);

  // ---- losses ----
  vq_sqdiff_reduce<<<4096, 256, 0, stream>>>(out, action, (size_t)4194304, accum + 0);
  vq_finalize_loss<<<1, 1, 0, stream>>>(accum, out + (out_size - 1));
}
